// LocalFeatureCorrelationLayer_24910810317042
// MI455X (gfx1250) — matrix operand order repaired and checked by reading, not compiled, not run on hardware
//
#include <hip/hip_runtime.h>
#include <hip/hip_bf16.h>
#include <math.h>

typedef __attribute__((ext_vector_type(16))) _Float16 v16h;
typedef __attribute__((ext_vector_type(8)))  _Float16 v8h;
typedef __attribute__((ext_vector_type(4)))  _Float16 v4h;
typedef __attribute__((ext_vector_type(8)))  float    v8f;
typedef __attribute__((ext_vector_type(4)))  float    v4f;

#define BATCH 8
#define CHN   256
#define HH    96
#define WW    96
#define PATCH 9
#define PADR  4
#define OC    81          // 9*9 correlation channels
#define WT    16          // w-tile (GEMM M)
#define NCOL  32          // source columns incl. halo (two 16-wide N tiles)
#define CPAD  264         // 256 + 8 f16 pad: rows stay 16B-aligned, bank stride 4
#define NTHREADS 288      // 9 wave32's, one per dy

__global__ __launch_bounds__(NTHREADS)
void local_corr_wmma(const float* __restrict__ src,
                     const float* __restrict__ tgt,
                     float* __restrict__ out)
{
    __shared__ _Float16 sA[WT][CPAD];            //  8.4 KB  target tile  [m][c]
    __shared__ _Float16 sB[PATCH][NCOL][CPAD];   // 152.1 KB source tiles [dy][n][c]
    __shared__ float    sCorr[OC][WT];           //  5.2 KB  relu'd band
    __shared__ float    sInv[WT];

    const int tid = threadIdx.x;
    const int blk = blockIdx.x;                  // b*H*(W/16) + h*(W/16) + wt
    const int wt  = blk % (WW / WT);
    const int h   = (blk / (WW / WT)) % HH;
    const int b   = blk / ((WW / WT) * HH);
    const int w0  = wt * WT;

    const size_t plane = (size_t)HH * WW;
    const float* tgtB = tgt + (size_t)b * CHN * plane;
    const float* srcB = src + (size_t)b * CHN * plane;

    // ---- Stage A (target): 4x4 (w x c) patches, b128 loads -> b64 LDS stores
    // tgt[c][h][w0+m] -> sA[m][c]
    for (int idx = tid; idx < (WT / 4) * (CHN / 4); idx += NTHREADS) {
        int m4 = (idx & 3) * 4;                  // w group (always in-bounds)
        int c4 = (idx >> 2) * 4;                 // c group
        v4f r[4];
        #pragma unroll
        for (int cc = 0; cc < 4; ++cc)
            r[cc] = *(const v4f*)&tgtB[(size_t)(c4 + cc) * plane + (size_t)h * WW + (w0 + m4)];
        #pragma unroll
        for (int mm = 0; mm < 4; ++mm) {
            v4h p;
            #pragma unroll
            for (int cc = 0; cc < 4; ++cc) p[cc] = (_Float16)r[cc][mm];
            *(v4h*)&sA[m4 + mm][c4] = p;         // 8B contiguous along c
        }
    }

    // ---- Stage B (source rows h+dy-4, cols w0-4..w0+27): 4x4 patches -------
    // src[c][h+dy-4][w0-4+n] -> sB[dy][n][c];  halo 4-groups are fully in/out
    for (int idx = tid; idx < PATCH * (NCOL / 4) * (CHN / 4); idx += NTHREADS) {
        int n4 = (idx & 7) * 4;                  // w group within the 32 cols
        int t  = idx >> 3;
        int c4 = (t & 63) * 4;                   // c group
        int dy = t >> 6;                         // 0..8
        int hs = h + dy - PADR;
        int ws = w0 - PADR + n4;                 // 4-aligned: group all-in or all-out
        bool ok = ((unsigned)hs < (unsigned)HH) && ((unsigned)ws < (unsigned)WW);
        v4f r[4];
        #pragma unroll
        for (int cc = 0; cc < 4; ++cc) {
            v4f z = {0.0f, 0.0f, 0.0f, 0.0f};
            r[cc] = ok ? *(const v4f*)&srcB[(size_t)(c4 + cc) * plane + (size_t)hs * WW + ws]
                       : z;
        }
        #pragma unroll
        for (int nn = 0; nn < 4; ++nn) {
            v4h p;
            #pragma unroll
            for (int cc = 0; cc < 4; ++cc) p[cc] = (_Float16)r[cc][nn];
            *(v4h*)&sB[dy][n4 + nn][c4] = p;     // 8B contiguous along c
        }
    }
    __syncthreads();

    // ---- One wave per dy: P(16x32) = A(16x256) x B(256x32) via WMMA --------
    const int dy   = tid >> 5;       // wave id == dy  (exactly 9 waves)
    const int lane = tid & 31;
    const int r16  = lane & 15;      // A row (M) / B column within N tile
    const int half = lane >> 4;      // K-half selector per ISA fragment layout

    #pragma unroll
    for (int nt = 0; nt < 2; ++nt) {
        v8f acc = {};
        const int ncol = nt * 16 + r16;          // global product column 0..31
        #pragma unroll
        for (int kc = 0; kc < CHN / 32; ++kc) {  // 8 K-chunks of 32
            const int kb = kc * 32;
            // A fragment: lane holds K = {kb+8h .. +7} and {kb+8h+16 .. +23}
            v8h a0 = *(const v8h*)&sA[r16][kb + half * 8];
            v8h a1 = *(const v8h*)&sA[r16][kb + half * 8 + 16];
            v16h a;
            #pragma unroll
            for (int i = 0; i < 8; ++i) { a[i] = a0[i]; a[8 + i] = a1[i]; }
            // B fragment: lane holds K = {kb+8h .. +7} and {kb+8h+16 .. +23} of its column
            v8h b0 = *(const v8h*)&sB[dy][ncol][kb + half * 8];
            v8h b1 = *(const v8h*)&sB[dy][ncol][kb + half * 8 + 16];
            v16h bf;
            #pragma unroll
            for (int i = 0; i < 8; ++i) { bf[i] = b0[i]; bf[8 + i] = b1[i]; }
            acc = __builtin_amdgcn_wmma_f32_16x16x32_f16(
                      false, a, false, bf, (short)0, acc, false, false);
        }
        // Extract band: corr[m][dx] = P[m][m+dx]; this lane owns column `ncol`.
        // D layout: lane(N=ncol) vgpr r holds M = r + 8*half.
        #pragma unroll
        for (int r = 0; r < 8; ++r) {
            int M  = r + half * 8;
            int dx = ncol - M;
            if (dx >= 0 && dx < PATCH) {
                float v = acc[r];
                sCorr[dy * PATCH + dx][M] = v > 0.0f ? v : 0.0f;  // ReLU
            }
        }
    }
    __syncthreads();

    // ---- L2 norm over the 81 channels per position -------------------------
    if (tid < WT) {
        float s = 0.0f;
        #pragma unroll
        for (int o = 0; o < OC; ++o) { float v = sCorr[o][tid]; s += v * v; }
        sInv[tid] = 1.0f / fmaxf(sqrtf(s), 1e-12f);
    }
    __syncthreads();

    // ---- Normalized, coalesced fp32 stores: out[b][o][h][w0+m] -------------
    float* outB = out + (size_t)b * OC * plane + (size_t)h * WW + w0;
    for (int idx = tid; idx < OC * WT; idx += NTHREADS) {
        int m = idx & (WT - 1);
        int o = idx >> 4;
        outB[(size_t)o * plane + m] = sCorr[o][m] * sInv[m];
    }
}

extern "C" void kernel_launch(void* const* d_in, const int* in_sizes, int n_in,
                              void* d_out, int out_size, void* d_ws, size_t ws_size,
                              hipStream_t stream)
{
    const float* src = (const float*)d_in[0];   // feature_source
    const float* tgt = (const float*)d_in[1];   // feature_target
    float* out = (float*)d_out;

    dim3 grid(BATCH * HH * (WW / WT));          // 8*96*6 = 4608 workgroups
    dim3 block(NTHREADS);                       // 9 wave32's
    local_corr_wmma<<<grid, block, 0, stream>>>(src, tgt, out);
}
